// JointConvLayer_56023553409097
// MI455X (gfx1250) — matrix operand order repaired and checked by reading, not compiled, not run on hardware
//
#include <hip/hip_runtime.h>

// ============================================================================
// CDNA5 (gfx1250) implementation of JointConvLayer forward.
//
// Strategy (MI455X):
//  * All dense GEMMs run through v_wmma_f32_16x16x32_bf16 (wave32, f32 accum).
//  * A-tiles are DMA'd into LDS by the Tensor Data Mover (tensor_load_to_lds,
//    TENSORcnt + s_wait_tensorcnt), using D# LDS-padding to produce the
//    bank-conflict-free stride-40 row layout directly.
//  * Per-edge weight tensor w (51200x1600, 327MB fp32) is never materialized:
//    produced/consumed in 10240-edge bf16 chunks (33MB) that stay in the
//    192MB L2.
//  * Edge-transition MLP (131072x384 @ 384x384 x2 + 384x128, ~90 GFLOP) is
//    chunked at 16384 rows so eh/th/sum intermediates (12.6MB bf16 each)
//    chain through L2.
//  * All GEMM callers guarantee M%64==0 and N%64==0 -> zero-guard staging.
// ============================================================================

typedef __bf16 bf16_t;
typedef __attribute__((ext_vector_type(16))) __bf16 v16bf;
typedef __attribute__((ext_vector_type(8)))  __bf16 v8bf;
typedef __attribute__((ext_vector_type(8)))  float  v8f;
typedef __attribute__((ext_vector_type(4)))  unsigned int v4u32;
typedef __attribute__((ext_vector_type(8)))  int v8i32;
typedef __attribute__((ext_vector_type(4)))  int v4i32;

#if defined(__HIP_DEVICE_COMPILE__) && __has_builtin(__builtin_amdgcn_tensor_load_to_lds)
#define USE_TDM 1
#else
#define USE_TDM 0
#endif

#define NS_   32
#define NV_   8
#define DIM_  56
#define WNUM_ 1600
#define BB_   2
#define LL_   256
#define NRES_ 512
#define NATOM_ 5120
#define EB_   10240
#define ER_   40960
#define EE_   51200
#define HE_   64
#define CS_   256
#define CZ_   128
#define CH_   16
#define HH_   16
#define PQ_   8
#define PV_   12
#define NEB_  128
#define HID_  384

#define OUT_DENSE_OFF 286720           // NATOM_*DIM_
#define OUT_EDGE_OFF  417792           // + NRES_*CS_

#define GFLAG_RELU  1
#define GFLAG_OUTBF 2

// ---------------------------------------------------------------------------
// WMMA bf16 GEMM:  D[M,N] = epilogue(A[M,K] @ B[K,N] + bias)
// block = 128 threads (4 waves), 64x64 tile, K stepped by 32.
// REQUIRES: M % 64 == 0, N % 64 == 0, K % 32 == 0 (all callers comply).
// ---------------------------------------------------------------------------
template <int FLAGS>
__global__ __launch_bounds__(128)
void k_wmma_gemm(const bf16_t* __restrict__ A, const bf16_t* __restrict__ Bw,
                 const float* __restrict__ bias, const bf16_t* __restrict__ resid,
                 float* __restrict__ Df, bf16_t* __restrict__ Dh,
                 int M, int N, int K)
{
    // single shared block so LDS offsets are known constants (TDM descriptor
    // takes a raw LDS byte address): lA at 0, lB at 5120.
    __shared__ bf16_t lsm[64 * 40 * 2];
    bf16_t* lA = lsm;            // [row][k], stride 40 (16B-aligned rows + pad)
    bf16_t* lB = lsm + 64 * 40;  // [n][k]   (transposed on store)

    const int t    = threadIdx.x;
    const int lane = t & 31;
    const int wave = t >> 5;
    const int wm   = (wave >> 1) * 32;
    const int wn   = (wave & 1) * 32;
    const int bm   = blockIdx.y * 64;
    const int bn   = blockIdx.x * 64;
    const int r    = lane & 15;
    const int hf   = lane >> 4;

    v8f acc[2][2];
    for (int i = 0; i < 2; ++i)
        for (int j = 0; j < 2; ++j)
            acc[i][j] = (v8f){0.f,0.f,0.f,0.f,0.f,0.f,0.f,0.f};

    for (int k0 = 0; k0 < K; k0 += 32) {
#if USE_TDM
        // ---- A tile (64 rows x 32 bf16) via Tensor Data Mover ----
        // D# group0: count=1 | lds_addr=0 | global tile addr | type=2
        // D# group1: data_size=2B, pad_enable, pad_interval=16dw (one row),
        //            pad_amount=4dw (8 bf16) -> stride-40 rows in LDS.
        if (t < 32) {
            unsigned long long ga =
                (unsigned long long)A + ((unsigned long long)bm * K + k0) * 2ull;
            v4u32 g0;
            g0.x = 1u;                                   // count=1, load, no gather
            g0.y = 0u;                                   // lds_addr = 0 (lA)
            g0.z = (unsigned)(ga & 0xFFFFFFFFull);       // global_addr[31:0]
            g0.w = (unsigned)((ga >> 32) & 0x01FFFFFFull) | (2u << 30); // [56:32]|type=2
            v8i32 g1;
            g1[0] = (1 << 16) | (1 << 20) | (3 << 22) | (3 << 25);
            g1[1] = (int)(((unsigned)K & 0xFFFFu) << 16);            // tensor_dim0 lo
            g1[2] = (int)((((unsigned)K >> 16) & 0xFFFFu) |
                          (((unsigned)M & 0xFFFFu) << 16));          // dim0 hi | dim1 lo
            g1[3] = (int)((((unsigned)M >> 16) & 0xFFFFu) | (32u << 16)); // dim1 hi | tile_dim0
            g1[4] = 64;                                  // tile_dim1=64, tile_dim2=0
            g1[5] = K;                                   // tensor_dim0_stride lo32 (elems)
            g1[6] = 0;
            g1[7] = 0;
            v4i32 gz = {0, 0, 0, 0};
#if defined(__clang_major__) && __clang_major__ >= 23
            v8i32 gz8 = {0, 0, 0, 0, 0, 0, 0, 0};
            __builtin_amdgcn_tensor_load_to_lds(g0, g1, gz, gz, gz8, 0);
#else
            __builtin_amdgcn_tensor_load_to_lds(g0, g1, gz, gz, 0);
#endif
        }
#else
        // ---- manual A staging (guard-free: M%64==0, K%32==0) ----
        for (int i = 0; i < 2; ++i) {
            int c   = t + i * 128;
            int row = c >> 2;
            int q   = (c & 3) * 8;
            *(uint4*)(lA + row * 40 + q) =
                *(const uint4*)(A + (size_t)(bm + row) * K + k0 + q);
        }
#endif
        // ---- B tile (32 x 64), coalesced load + transpose into [n][k] ----
        for (int i = 0; i < 2; ++i) {
            int c  = t + i * 128;
            int kk = c >> 3;
            int n0 = (c & 7) * 8;
            bf16_t tmp[8];
            *(uint4*)tmp = *(const uint4*)(Bw + (size_t)(k0 + kk) * N + bn + n0);
            for (int j = 0; j < 8; ++j) lB[(n0 + j) * 40 + kk] = tmp[j];
        }
        // prefetch next K slab of B (global_prefetch_b8)
        if (k0 + 32 < K) {
            int kk = t >> 3;
            __builtin_prefetch(Bw + (size_t)(k0 + 32 + kk) * N + bn, 0, 1);
        }
#if USE_TDM
        if (t < 32) __builtin_amdgcn_s_wait_tensorcnt(0);
#endif
        __syncthreads();

        // ---- fragments per ISA 16-bit A/B lane layouts ----
        v16bf afrag[2], bfrag[2];
        for (int fm = 0; fm < 2; ++fm) {
            int row = wm + fm * 16 + r;
            union { v16bf v; v8bf h[2]; } u;
            u.h[0] = *(const v8bf*)(lA + row * 40 + hf * 8);
            u.h[1] = *(const v8bf*)(lA + row * 40 + 16 + hf * 8);
            afrag[fm] = u.v;
        }
        for (int fn = 0; fn < 2; ++fn) {
            int col = wn + fn * 16 + r;
            union { v16bf v; v8bf h[2]; } u;
            const v8bf* p = (const v8bf*)(lB + col * 40 + hf * 8);
            u.h[0] = p[0];
            u.h[1] = p[2];
            bfrag[fn] = u.v;
        }
        for (int fm = 0; fm < 2; ++fm)
            for (int fn = 0; fn < 2; ++fn)
                acc[fm][fn] = __builtin_amdgcn_wmma_f32_16x16x32_bf16(
                    false, afrag[fm], false, bfrag[fn],
                    (short)0, acc[fm][fn], false, false);
        __syncthreads();
    }

    // ---- epilogue: bias, relu, residual, f32/bf16 store (guard-free) ----
    for (int fm = 0; fm < 2; ++fm)
        for (int fn = 0; fn < 2; ++fn) {
            int col = bn + wn + fn * 16 + r;
            float bv = bias ? bias[col] : 0.0f;
            for (int i = 0; i < 8; ++i) {
                int row = bm + wm + fm * 16 + hf * 8 + i;
                float v = acc[fm][fn][i] + bv;
                if (FLAGS & GFLAG_RELU) v = fmaxf(v, 0.0f);
                if (resid) v += (float)resid[(size_t)row * N + col];
                if (FLAGS & GFLAG_OUTBF) Dh[(size_t)row * N + col] = (bf16_t)v;
                else                     Df[(size_t)row * N + col] = v;
            }
        }
}

// ---------------------------------------------------------------------------
// Elementwise / utility kernels
// ---------------------------------------------------------------------------
__global__ void k_cvt(const float* __restrict__ x, bf16_t* __restrict__ y, long long n)
{
    long long i = (long long)blockIdx.x * 256 + threadIdx.x;
    if (i < n) y[i] = (bf16_t)x[i];
}

// pad K(rows) dim of weight [rows,cols] -> [rowsp,cols] bf16
__global__ void k_cvt_padrows(const float* __restrict__ x, bf16_t* __restrict__ y,
                              int rows, int rowsp, int cols)
{
    long long i = (long long)blockIdx.x * 256 + threadIdx.x;
    if (i >= (long long)rowsp * cols) return;
    int rr = (int)(i / cols);
    y[i] = (rr < rows) ? (bf16_t)x[i] : (bf16_t)0.0f;
}

__global__ void k_count(const int* __restrict__ idx, float* __restrict__ cnt, int n)
{
    int i = blockIdx.x * 256 + threadIdx.x;
    if (i < n) atomicAdd(&cnt[idx[i]], 1.0f);
}

// tensor product message + scatter-add.  thread = (edge_local, k<56)
__global__ void k_tp(const float* __restrict__ feat, const int* __restrict__ src,
                     const int* __restrict__ dstIdx, const float* __restrict__ sh,
                     const bf16_t* __restrict__ w, int e0, int ne,
                     float* __restrict__ accum)
{
    long long i = (long long)blockIdx.x * 256 + threadIdx.x;
    if (i >= (long long)ne * 56) return;
    int el = (int)(i / 56), k = (int)(i % 56);
    int e  = e0 + el;
    int fr = src ? src[e] : e;
    const float* fs = feat + (size_t)fr * DIM_;
    const float* se = sh + (size_t)e * 4;
    const bf16_t* we = w + (size_t)el * WNUM_;
    const float y0 = se[0];
    const float is32 = 0.17677669529663687f;   // 1/sqrt(32)
    const float is8  = 0.35355339059327373f;   // 1/sqrt(8)
    const float isq2 = 0.70710678118654752f;
    const float isq3 = 0.57735026918962576f;
    float out;
    if (k < NS_) {
        float o1 = 0.f;
        for (int s = 0; s < NS_; ++s)
            o1 += fs[s] * y0 * (float)we[s * 32 + k];
        float o2 = 0.f;
        for (int v = 0; v < NV_; ++v) {
            float d = (fs[32 + v*3 + 0] * se[1] + fs[32 + v*3 + 1] * se[2] +
                       fs[32 + v*3 + 2] * se[3]) * isq3;
            o2 += d * (float)we[1024 + v * 32 + k];
        }
        out = (o1 * is32 + o2 * is8) * isq2;
    } else {
        int kv = (k - 32) / 3, c = (k - 32) % 3;
        float o1 = 0.f;
        for (int s = 0; s < NS_; ++s)
            o1 += fs[s] * (float)we[1280 + s * 8 + kv];
        o1 *= se[1 + c] * is32;
        float o2 = 0.f;
        for (int v = 0; v < NV_; ++v)
            o2 += fs[32 + v*3 + c] * (float)we[1536 + v * 8 + kv];
        o2 *= y0 * is8;
        out = (o1 + o2) * isq2;
    }
    atomicAdd(&accum[(size_t)dstIdx[e] * 56 + k], out);
}

__global__ void k_atom_out(const float* __restrict__ feat, const float* __restrict__ acc,
                           const float* __restrict__ cnt, float* __restrict__ out, int n)
{
    long long i = (long long)blockIdx.x * 256 + threadIdx.x;
    if (i >= (long long)n * 56) return;
    int a = (int)(i / 56);
    out[i] = feat[i] + acc[i] / fmaxf(cnt[a], 1.0f);
}

__global__ void k_quat(const float* __restrict__ q4, float* __restrict__ R, int n)
{
    int i = blockIdx.x * 64 + threadIdx.x;
    if (i >= n) return;
    float w = q4[i*4], x = q4[i*4+1], y = q4[i*4+2], z = q4[i*4+3];
    float inv = rsqrtf(w*w + x*x + y*y + z*z);
    w *= inv; x *= inv; y *= inv; z *= inv;
    float* r = R + (size_t)i * 9;
    r[0] = 1.f - 2.f*(y*y + z*z); r[1] = 2.f*(x*y - w*z); r[2] = 2.f*(x*z + w*y);
    r[3] = 2.f*(x*y + w*z); r[4] = 1.f - 2.f*(x*x + z*z); r[5] = 2.f*(y*z - w*x);
    r[6] = 2.f*(x*z - w*y); r[7] = 2.f*(y*z + w*x); r[8] = 1.f - 2.f*(x*x + y*y);
}

// res_update: mean + vector rotation, write padded bf16 [NRES,64]
__global__ void k_resupd(const float* __restrict__ acc, const float* __restrict__ cnt,
                         const float* __restrict__ R, bf16_t* __restrict__ outp, int n)
{
    long long i = (long long)blockIdx.x * 256 + threadIdx.x;
    if (i >= (long long)n * 64) return;
    int nr = (int)(i >> 6), c = (int)(i & 63);
    float cm = fmaxf(cnt[nr], 1.0f);
    float val = 0.f;
    if (c < NS_) {
        val = acc[(size_t)nr * 56 + c] / cm;
    } else if (c < 56) {
        int vv = (c - 32) / 3, cc = (c - 32) % 3;
        for (int d = 0; d < 3; ++d)   // vv_c = sum_d R[d][c] * v_d
            val += R[(size_t)nr * 9 + d * 3 + cc] * acc[(size_t)nr * 56 + 32 + vv*3 + d];
        val /= cm;
    }
    outp[i] = (bf16_t)val;
}

// layernorm over last dim C; optional residual add first; dual f32/bf16 out
__global__ __launch_bounds__(128)
void k_ln(const float* __restrict__ x, const float* __restrict__ resid,
          const float* __restrict__ g, const float* __restrict__ bb,
          float* __restrict__ out, bf16_t* __restrict__ outbf, int C)
{
    __shared__ float red[128];
    int row = blockIdx.x, t = threadIdx.x;
    const float* xr = x + (size_t)row * C;
    const float* rr = resid ? resid + (size_t)row * C : nullptr;
    float s = 0.f;
    for (int c = t; c < C; c += 128) s += xr[c] + (rr ? rr[c] : 0.f);
    red[t] = s; __syncthreads();
    for (int o = 64; o > 0; o >>= 1) { if (t < o) red[t] += red[t + o]; __syncthreads(); }
    float mean = red[0] / C; __syncthreads();
    float s2 = 0.f;
    for (int c = t; c < C; c += 128) {
        float v = xr[c] + (rr ? rr[c] : 0.f) - mean; s2 += v * v;
    }
    red[t] = s2; __syncthreads();
    for (int o = 64; o > 0; o >>= 1) { if (t < o) red[t] += red[t + o]; __syncthreads(); }
    float rstd = rsqrtf(red[0] / C + 1e-5f);
    for (int c = t; c < C; c += 128) {
        float v = (xr[c] + (rr ? rr[c] : 0.f) - mean) * rstd * g[c] + bb[c];
        if (out)   out[(size_t)row * C + c] = v;
        if (outbf) outbf[(size_t)row * C + c] = (bf16_t)v;
    }
}

__global__ void k_softplus(const float* __restrict__ g, float* __restrict__ o, int n)
{
    int i = threadIdx.x;
    if (i < n) o[i] = log1pf(expf(g[i]));
}

// local points -> global: p_g = R p + t.  thread = (res, point)
__global__ void k_points(const float* __restrict__ pl, const float* __restrict__ R,
                         const float* __restrict__ tr, float* __restrict__ pg,
                         int npts, int n)
{
    long long i = (long long)blockIdx.x * 256 + threadIdx.x;
    if (i >= (long long)n * npts) return;
    int nr = (int)(i / npts), pt = (int)(i % npts);
    const float* Rn = R + (size_t)nr * 9;
    const float* p  = pl + ((size_t)nr * npts + pt) * 3;
    const float* t  = tr + (size_t)nr * 3;
    float* o = pg + ((size_t)nr * npts + pt) * 3;
    for (int c = 0; c < 3; ++c)
        o[c] = Rn[c*3+0]*p[0] + Rn[c*3+1]*p[1] + Rn[c*3+2]*p[2] + t[c];
}

__global__ void k_norms(const float* __restrict__ qpg, const float* __restrict__ kvpg,
                        float* __restrict__ qn, float* __restrict__ kn, int n)
{
    int i = blockIdx.x * 256 + threadIdx.x;
    if (i >= n * HH_) return;
    int nr = i >> 4, h = i & 15;
    float sq = 0.f, sk = 0.f;
    for (int p = 0; p < PQ_; ++p)
        for (int c = 0; c < 3; ++c) {
            float a = qpg[(size_t)nr * (HH_*PQ_*3) + (h*PQ_ + p)*3 + c];
            sq += a * a;
            float b = kvpg[(size_t)nr * (HH_*(PQ_+PV_)*3) + (h*(PQ_+PV_) + p)*3 + c];
            sk += b * b;
        }
    qn[i] = sq; kn[i] = sk;
}

// bz[r,h] = z[r,:] . wb[:,h] + bb[h]   (N=16: plain kernel beats a WMMA tile)
__global__ void k_bz(const float* __restrict__ z, const float* __restrict__ wb,
                     const float* __restrict__ bbias, float* __restrict__ bz)
{
    long long i = (long long)blockIdx.x * 256 + threadIdx.x;
    if (i >= (long long)BB_ * LL_ * LL_ * HH_) return;
    long long rr = i >> 4; int h = (int)(i & 15);
    const float* zr = z + rr * CZ_;
    float s = bbias[h];
    for (int c = 0; c < CZ_; ++c) s += zr[c] * wb[c * HH_ + h];
    bz[i] = s;
}

// attention logits, layout att[b,h,i,j]
__global__ void k_att(const float* __restrict__ q, const float* __restrict__ kv,
                      const float* __restrict__ qpg, const float* __restrict__ kvpg,
                      const float* __restrict__ qn, const float* __restrict__ kn,
                      const float* __restrict__ bz, const float* __restrict__ gsp,
                      const float* __restrict__ mask, float* __restrict__ att)
{
    long long i = (long long)blockIdx.x * 256 + threadIdx.x;
    if (i >= (long long)BB_ * HH_ * LL_ * LL_) return;
    int j  = (int)(i & 255);
    int ii = (int)((i >> 8) & 255);
    int h  = (int)((i >> 16) & 15);
    int b  = (int)(i >> 20);
    int ni = b * LL_ + ii, nj = b * LL_ + j;
    const float wL = 0.57735026918962576f;
    float dotqk = 0.f;
    for (int c = 0; c < CH_; ++c)
        dotqk += q[(size_t)ni * 256 + h*16 + c] * kv[(size_t)nj * 512 + h*32 + c];
    float a = dotqk * (wL * 0.25f);                     // wL / sqrt(CH)
    a += wL * bz[((size_t)ni * LL_ + j) * HH_ + h];
    float dpt = 0.f;
    for (int p = 0; p < PQ_; ++p)
        for (int c = 0; c < 3; ++c)
            dpt += qpg[(size_t)ni * 384 + (h*8 + p)*3 + c] *
                   kvpg[(size_t)nj * 960 + (h*20 + p)*3 + c];
    float d2 = qn[ni*16 + h] + kn[nj*16 + h] - 2.f * dpt;
    const float wC = 0.16666666666666666f;              // sqrt(2/(9*PQ))
    a -= 0.5f * wC * wL * gsp[h] * d2;
    a += (mask[ni] * mask[nj] - 1.f) * 1e9f;
    att[i] = a;
}

__global__ __launch_bounds__(256)
void k_softmax(float* __restrict__ att)
{
    __shared__ float red[256];
    size_t row = blockIdx.x; int t = threadIdx.x;
    float v = att[row * 256 + t];
    red[t] = v; __syncthreads();
    for (int o = 128; o > 0; o >>= 1) { if (t < o) red[t] = fmaxf(red[t], red[t+o]); __syncthreads(); }
    float mx = red[0]; __syncthreads();
    float e = __expf(v - mx);
    red[t] = e; __syncthreads();
    for (int o = 128; o > 0; o >>= 1) { if (t < o) red[t] += red[t+o]; __syncthreads(); }
    att[row * 256 + t] = e / red[0];
}

// o = a @ v  -> cat[:, 0:256]
__global__ void k_oout(const float* __restrict__ att, const float* __restrict__ kv,
                       bf16_t* __restrict__ catbf)
{
    int i = blockIdx.x * 256 + threadIdx.x;
    if (i >= NRES_ * 256) return;
    int n = i >> 8, hc = i & 255;
    int h = hc >> 4, c = hc & 15;
    int b = n >> 8, ii = n & 255;
    const float* ar = att + (((size_t)(b*HH_ + h) * LL_) + ii) * LL_;
    float s = 0.f;
    for (int j = 0; j < LL_; ++j)
        s += ar[j] * kv[((size_t)(b*LL_ + j)) * 512 + h*32 + 16 + c];
    catbf[(size_t)n * 3072 + hc] = (bf16_t)s;
}

__global__ void k_opt(const float* __restrict__ att, const float* __restrict__ kvpg,
                      float* __restrict__ optg)
{
    int i = blockIdx.x * 256 + threadIdx.x;
    if (i >= NRES_ * 576) return;
    int n = i / 576, idx = i % 576;       // idx = (h*12+p)*3 + c
    int c = idx % 3, hp = idx / 3;
    int h = hp / 12, p = hp % 12;
    int b = n >> 8, ii = n & 255;
    const float* ar = att + (((size_t)(b*HH_ + h) * LL_) + ii) * LL_;
    float s = 0.f;
    for (int j = 0; j < LL_; ++j)
        s += ar[j] * kvpg[((size_t)(b*LL_ + j)) * 960 + (h*20 + 8 + p)*3 + c];
    optg[(size_t)n * 576 + idx] = s;
}

// back to local frame + norm -> cat[:, 256:832] and cat[:, 832:1024]
__global__ void k_optfin(const float* __restrict__ optg, const float* __restrict__ R,
                         const float* __restrict__ tr, bf16_t* __restrict__ catbf)
{
    int i = blockIdx.x * 256 + threadIdx.x;
    if (i >= NRES_ * 192) return;
    int n = i / 192, hp = i % 192;
    const float* Rn = R + (size_t)n * 9;
    const float* t  = tr + (size_t)n * 3;
    float x[3], loc[3];
    for (int d = 0; d < 3; ++d) x[d] = optg[(size_t)n * 576 + hp*3 + d] - t[d];
    float nn = 1e-8f;
    for (int c = 0; c < 3; ++c) {
        loc[c] = Rn[0*3+c]*x[0] + Rn[1*3+c]*x[1] + Rn[2*3+c]*x[2];  // R^T (x - t)
        nn += loc[c] * loc[c];
        catbf[(size_t)n * 3072 + 256 + hp*3 + c] = (bf16_t)loc[c];
    }
    catbf[(size_t)n * 3072 + 832 + hp] = (bf16_t)sqrtf(nn);
}

// opair = a @ z -> cat[:, 1024:3072]
__global__ void k_opair(const float* __restrict__ att, const float* __restrict__ zin,
                        bf16_t* __restrict__ catbf)
{
    long long i = (long long)blockIdx.x * 256 + threadIdx.x;
    if (i >= (long long)NRES_ * 2048) return;
    int n = (int)(i >> 11), idx = (int)(i & 2047);
    int h = idx >> 7, zz = idx & 127;
    int b = n >> 8, ii = n & 255;
    const float* ar = att + (((size_t)(b*HH_ + h) * LL_) + ii) * LL_;
    const float* zr = zin + (((size_t)(b*LL_ + ii)) * LL_) * CZ_ + zz;
    float s = 0.f;
    for (int j = 0; j < LL_; ++j) s += ar[j] * zr[(size_t)j * CZ_];
    catbf[(size_t)n * 3072 + 1024 + idx] = (bf16_t)s;
}

// eh chunk builder: [z | ne_i | ne_j] -> bf16
__global__ void k_build_eh(const float* __restrict__ zin, const float* __restrict__ ne,
                           bf16_t* __restrict__ eh, long long r0, int nrows)
{
    long long i = (long long)blockIdx.x * 256 + threadIdx.x;
    if (i >= (long long)nrows * HID_) return;
    int c = (int)(i % HID_);
    long long rg = r0 + i / HID_;
    int j  = (int)(rg & 255);
    int ii = (int)((rg >> 8) & 255);
    int b  = (int)(rg >> 16);
    float v;
    if (c < CZ_)           v = zin[rg * CZ_ + c];
    else if (c < CZ_+NEB_) v = ne[((size_t)(b*LL_ + ii)) * NEB_ + (c - CZ_)];
    else                   v = ne[((size_t)(b*LL_ + j))  * NEB_ + (c - CZ_ - NEB_)];
    eh[i] = (bf16_t)v;
}

// ---------------------------------------------------------------------------
// Host side
// ---------------------------------------------------------------------------
static inline void gemm(hipStream_t s, const bf16_t* A, const bf16_t* Bw,
                        const float* bias, const bf16_t* resid, void* D,
                        int M, int N, int K, int flags)
{
    dim3 g((unsigned)(N / 64), (unsigned)(M / 64));
    float*  Df = (flags & GFLAG_OUTBF) ? nullptr : (float*)D;
    bf16_t* Dh = (flags & GFLAG_OUTBF) ? (bf16_t*)D : nullptr;
    switch (flags) {
    case GFLAG_RELU | GFLAG_OUTBF:
        k_wmma_gemm<GFLAG_RELU | GFLAG_OUTBF><<<g, 128, 0, s>>>(A, Bw, bias, resid, Df, Dh, M, N, K);
        break;
    case GFLAG_OUTBF:
        k_wmma_gemm<GFLAG_OUTBF><<<g, 128, 0, s>>>(A, Bw, bias, resid, Df, Dh, M, N, K);
        break;
    default:
        k_wmma_gemm<0><<<g, 128, 0, s>>>(A, Bw, bias, resid, Df, Dh, M, N, K);
        break;
    }
}

static inline void cvt(hipStream_t s, const float* x, bf16_t* y, long long n)
{
    k_cvt<<<dim3((unsigned)((n + 255) / 256)), 256, 0, s>>>(x, y, n);
}

extern "C" void kernel_launch(void* const* d_in, const int* in_sizes, int n_in,
                              void* d_out_, int out_size, void* d_ws, size_t ws_size,
                              hipStream_t stream)
{
    float* d_out = (float*)d_out_;
    // ---- inputs (setup_inputs dict order) ----
    const float* atom_features = (const float*)d_in[0];
    const int*   edge_idx      = (const int*)d_in[1];     // (2, E)
    const float* bond_feat     = (const float*)d_in[2];
    const float* rad_feat      = (const float*)d_in[3];
    const float* edge_sh       = (const float*)d_in[4];
    const float* res_features  = (const float*)d_in[5];
    const float* agg_feat      = (const float*)d_in[6];
    const float* agg_sh        = (const float*)d_in[7];
    const int*   res_batch     = (const int*)d_in[8];
    const float* quats         = (const float*)d_in[9];
    const float* trans         = (const float*)d_in[10];
    const float* zin           = (const float*)d_in[11];
    const float* mask          = (const float*)d_in[12];
    // params, insertion order
    const float *p_bond_w1=(const float*)d_in[13], *p_bond_b1=(const float*)d_in[14];
    const float *p_bond_w2=(const float*)d_in[15], *p_bond_b2=(const float*)d_in[16];
    const float *p_rad_w1 =(const float*)d_in[17], *p_rad_b1 =(const float*)d_in[18];
    const float *p_rad_w2 =(const float*)d_in[19], *p_rad_b2 =(const float*)d_in[20];
    const float *p_agg_w1 =(const float*)d_in[21], *p_agg_b1 =(const float*)d_in[22];
    const float *p_agg_w2 =(const float*)d_in[23], *p_agg_b2 =(const float*)d_in[24];
    const float *p_i2s_w1 =(const float*)d_in[25], *p_i2s_b1 =(const float*)d_in[26];
    const float *p_i2s_w2 =(const float*)d_in[27], *p_i2s_b2 =(const float*)d_in[28];
    const float *p_i2s_w3 =(const float*)d_in[29], *p_i2s_b3 =(const float*)d_in[30];
    const float *p_a2r_g  =(const float*)d_in[31], *p_a2r_b  =(const float*)d_in[32];
    const float *p_ipa_g  =(const float*)d_in[33], *p_ipa_b  =(const float*)d_in[34];
    const float *p_wq     =(const float*)d_in[35], *p_bq     =(const float*)d_in[36];
    const float *p_wkv    =(const float*)d_in[37], *p_bkv    =(const float*)d_in[38];
    const float *p_wqp    =(const float*)d_in[39], *p_bqp    =(const float*)d_in[40];
    const float *p_wkvp   =(const float*)d_in[41], *p_bkvp   =(const float*)d_in[42];
    const float *p_wb     =(const float*)d_in[43], *p_bb     =(const float*)d_in[44];
    const float *p_gamma  =(const float*)d_in[45];
    const float *p_wo     =(const float*)d_in[46], *p_bo     =(const float*)d_in[47];
    const float *p_et_wi  =(const float*)d_in[48], *p_et_bi  =(const float*)d_in[49];
    const float *p_et_w1  =(const float*)d_in[50], *p_et_b1  =(const float*)d_in[51];
    const float *p_et_w2  =(const float*)d_in[52], *p_et_b2  =(const float*)d_in[53];
    const float *p_et_wf  =(const float*)d_in[54], *p_et_bf  =(const float*)d_in[55];
    const float *p_et_g   =(const float*)d_in[56], *p_et_b   =(const float*)d_in[57];

    const int* edge_src = edge_idx;
    const int* edge_dst = edge_idx + EE_;

    // ---- workspace bump allocator (deterministic every call) ----
    char* wsb = (char*)d_ws; size_t wso = 0;
    auto walloc = [&](size_t bytes) -> void* {
        void* p = wsb + wso; wso += (bytes + 255) & ~(size_t)255; return p;
    };
    // bf16 weights
    bf16_t* wb_bond_w1 = (bf16_t*)walloc(64*64*2);
    bf16_t* wb_bond_w2 = (bf16_t*)walloc(64*1600*2);
    bf16_t* wb_rad_w1  = (bf16_t*)walloc(64*64*2);
    bf16_t* wb_rad_w2  = (bf16_t*)walloc(64*1600*2);
    bf16_t* wb_agg_w1  = (bf16_t*)walloc(64*64*2);
    bf16_t* wb_agg_w2  = (bf16_t*)walloc(64*1600*2);
    bf16_t* wb_i2s_w1  = (bf16_t*)walloc(64*256*2);    // K padded 56->64
    bf16_t* wb_i2s_w2  = (bf16_t*)walloc(256*256*2);
    bf16_t* wb_i2s_w3  = (bf16_t*)walloc(256*256*2);
    bf16_t* wb_wq      = (bf16_t*)walloc(256*256*2);
    bf16_t* wb_wkv     = (bf16_t*)walloc(256*512*2);
    bf16_t* wb_wqp     = (bf16_t*)walloc(256*384*2);
    bf16_t* wb_wkvp    = (bf16_t*)walloc(256*960*2);
    bf16_t* wb_wo      = (bf16_t*)walloc(3072*256*2);
    bf16_t* wb_et_wi   = (bf16_t*)walloc(256*128*2);
    bf16_t* wb_et_w1   = (bf16_t*)walloc(384*384*2);
    bf16_t* wb_et_w2   = (bf16_t*)walloc(384*384*2);
    bf16_t* wb_et_wf   = (bf16_t*)walloc(384*128*2);
    // atom stage
    bf16_t* xedge_bf = (bf16_t*)walloc((size_t)EE_*64*2);
    bf16_t* hedge_bf = (bf16_t*)walloc((size_t)EE_*64*2);
    bf16_t* xagg_bf  = (bf16_t*)walloc((size_t)NATOM_*64*2);
    bf16_t* hagg_bf  = (bf16_t*)walloc((size_t)NATOM_*64*2);
    bf16_t* wchunk   = (bf16_t*)walloc((size_t)10240*1600*2);   // 33MB, L2-resident
    float*  atomacc  = (float*)walloc((size_t)NATOM_*56*4);
    float*  atomcnt  = (float*)walloc((size_t)NATOM_*4);
    float*  resacc   = (float*)walloc((size_t)NRES_*56*4);
    float*  rescnt   = (float*)walloc((size_t)NRES_*4);
    float*  Rmat     = (float*)walloc((size_t)NRES_*9*4);
    // residue stage
    bf16_t* respad_bf = (bf16_t*)walloc((size_t)NRES_*64*2);
    bf16_t* h1_bf     = (bf16_t*)walloc((size_t)NRES_*256*2);
    bf16_t* h2_bf     = (bf16_t*)walloc((size_t)NRES_*256*2);
    float*  i2sout    = (float*)walloc((size_t)NRES_*256*4);
    float*  resfeats  = (float*)walloc((size_t)NRES_*256*4);
    bf16_t* s_bf      = (bf16_t*)walloc((size_t)NRES_*256*2);
    float*  qbuf      = (float*)walloc((size_t)NRES_*256*4);
    float*  kvbuf     = (float*)walloc((size_t)NRES_*512*4);
    float*  qpl       = (float*)walloc((size_t)NRES_*384*4);
    float*  kvpl      = (float*)walloc((size_t)NRES_*960*4);
    float*  qpg       = (float*)walloc((size_t)NRES_*384*4);
    float*  kvpg      = (float*)walloc((size_t)NRES_*960*4);
    float*  qn        = (float*)walloc((size_t)NRES_*16*4);
    float*  kn        = (float*)walloc((size_t)NRES_*16*4);
    float*  bz        = (float*)walloc((size_t)BB_*LL_*LL_*HH_*4);
    float*  gsp       = (float*)walloc(64);
    float*  att       = (float*)walloc((size_t)BB_*HH_*LL_*LL_*4);
    float*  optg      = (float*)walloc((size_t)NRES_*576*4);
    bf16_t* catbf     = (bf16_t*)walloc((size_t)NRES_*3072*2);
    float*  ipaout    = (float*)walloc((size_t)NRES_*256*4);
    bf16_t* densebf   = (bf16_t*)walloc((size_t)NRES_*256*2);
    float*  nebuf     = (float*)walloc((size_t)NRES_*128*4);
    // edge-transition chunks
    const int NCH = 16384;
    bf16_t* ehc  = (bf16_t*)walloc((size_t)NCH*HID_*2);
    bf16_t* thc  = (bf16_t*)walloc((size_t)NCH*HID_*2);
    bf16_t* sumc = (bf16_t*)walloc((size_t)NCH*HID_*2);
    float*  outc = (float*)walloc((size_t)NCH*CZ_*4);

    // ---- zero accumulators ----
    hipMemsetAsync(atomacc, 0, (size_t)NATOM_*56*4, stream);
    hipMemsetAsync(atomcnt, 0, (size_t)NATOM_*4, stream);
    hipMemsetAsync(resacc,  0, (size_t)NRES_*56*4, stream);
    hipMemsetAsync(rescnt,  0, (size_t)NRES_*4, stream);

    // ---- convert weights to bf16 ----
    cvt(stream, p_bond_w1, wb_bond_w1, 64*64);
    cvt(stream, p_bond_w2, wb_bond_w2, 64*1600);
    cvt(stream, p_rad_w1,  wb_rad_w1,  64*64);
    cvt(stream, p_rad_w2,  wb_rad_w2,  64*1600);
    cvt(stream, p_agg_w1,  wb_agg_w1,  64*64);
    cvt(stream, p_agg_w2,  wb_agg_w2,  64*1600);
    k_cvt_padrows<<<dim3((64*256+255)/256),256,0,stream>>>(p_i2s_w1, wb_i2s_w1, 56, 64, 256);
    cvt(stream, p_i2s_w2, wb_i2s_w2, 256*256);
    cvt(stream, p_i2s_w3, wb_i2s_w3, 256*256);
    cvt(stream, p_wq,   wb_wq,   256*256);
    cvt(stream, p_wkv,  wb_wkv,  256*512);
    cvt(stream, p_wqp,  wb_wqp,  256*384);
    cvt(stream, p_wkvp, wb_wkvp, 256*960);
    cvt(stream, p_wo,   wb_wo,   3072*256);
    cvt(stream, p_et_wi, wb_et_wi, 256*128);
    cvt(stream, p_et_w1, wb_et_w1, 384*384);
    cvt(stream, p_et_w2, wb_et_w2, 384*384);
    cvt(stream, p_et_wf, wb_et_wf, 384*128);

    // ================= Stage A: atom message passing =================
    cvt(stream, bond_feat, xedge_bf, (long long)EB_*64);
    cvt(stream, rad_feat,  xedge_bf + (size_t)EB_*64, (long long)ER_*64);
    gemm(stream, xedge_bf, wb_bond_w1, p_bond_b1, nullptr, hedge_bf,
         EB_, 64, 64, GFLAG_RELU | GFLAG_OUTBF);
    gemm(stream, xedge_bf + (size_t)EB_*64, wb_rad_w1, p_rad_b1, nullptr,
         hedge_bf + (size_t)EB_*64, ER_, 64, 64, GFLAG_RELU | GFLAG_OUTBF);

    k_count<<<dim3((EE_+255)/256),256,0,stream>>>(edge_dst, atomcnt, EE_);
    k_count<<<dim3((NATOM_+255)/256),256,0,stream>>>(res_batch, rescnt, NATOM_);

    for (int c = 0; c < 5; ++c) {                       // 10240-edge chunks
        int e0 = c * 10240;
        const bf16_t* w2 = (c == 0) ? wb_bond_w2 : wb_rad_w2;
        const float*  b2 = (c == 0) ? p_bond_b2  : p_rad_b2;
        gemm(stream, hedge_bf + (size_t)e0*64, w2, b2, nullptr, wchunk,
             10240, WNUM_, 64, GFLAG_OUTBF);
        k_tp<<<dim3((10240*56+255)/256),256,0,stream>>>(
            atom_features, edge_src, edge_dst, edge_sh, wchunk, e0, 10240, atomacc);
    }
    k_atom_out<<<dim3((NATOM_*56+255)/256),256,0,stream>>>(
        atom_features, atomacc, atomcnt, d_out /* atom_out */, NATOM_);

    // ================= Stage B: atom -> residue =================
    cvt(stream, agg_feat, xagg_bf, (long long)NATOM_*64);
    gemm(stream, xagg_bf, wb_agg_w1, p_agg_b1, nullptr, hagg_bf,
         NATOM_, 64, 64, GFLAG_RELU | GFLAG_OUTBF);
    gemm(stream, hagg_bf, wb_agg_w2, p_agg_b2, nullptr, wchunk,
         NATOM_, WNUM_, 64, GFLAG_OUTBF);
    k_tp<<<dim3((NATOM_*56+255)/256),256,0,stream>>>(
        d_out /* atom_out */, nullptr, res_batch, agg_sh, wchunk, 0, NATOM_, resacc);

    k_quat<<<dim3((NRES_+63)/64),64,0,stream>>>(quats, Rmat, NRES_);
    k_resupd<<<dim3((NRES_*64+255)/256),256,0,stream>>>(resacc, rescnt, Rmat, respad_bf, NRES_);

    // i2s MLP chain
    gemm(stream, respad_bf, wb_i2s_w1, p_i2s_b1, nullptr, h1_bf, NRES_, 256, 64,
         GFLAG_RELU | GFLAG_OUTBF);
    gemm(stream, h1_bf, wb_i2s_w2, p_i2s_b2, nullptr, h2_bf, NRES_, 256, 256,
         GFLAG_RELU | GFLAG_OUTBF);
    gemm(stream, h2_bf, wb_i2s_w3, p_i2s_b3, nullptr, i2sout, NRES_, 256, 256, 0);
    k_ln<<<dim3(NRES_),128,0,stream>>>(i2sout, res_features, p_a2r_g, p_a2r_b,
                                       resfeats, s_bf, CS_);

    // ================= Stage C: IPA =================
    gemm(stream, s_bf, wb_wq,   p_bq,   nullptr, qbuf,  NRES_, 256, 256, 0);
    gemm(stream, s_bf, wb_wkv,  p_bkv,  nullptr, kvbuf, NRES_, 512, 256, 0);
    gemm(stream, s_bf, wb_wqp,  p_bqp,  nullptr, qpl,   NRES_, 384, 256, 0);
    gemm(stream, s_bf, wb_wkvp, p_bkvp, nullptr, kvpl,  NRES_, 960, 256, 0);
    k_points<<<dim3((NRES_*128+255)/256),256,0,stream>>>(qpl,  Rmat, trans, qpg,  128, NRES_);
    k_points<<<dim3((NRES_*320+255)/256),256,0,stream>>>(kvpl, Rmat, trans, kvpg, 320, NRES_);
    k_norms<<<dim3((NRES_*16+255)/256),256,0,stream>>>(qpg, kvpg, qn, kn, NRES_);
    k_bz<<<dim3((unsigned)(((long long)BB_*LL_*LL_*HH_+255)/256)),256,0,stream>>>(zin, p_wb, p_bb, bz);
    k_softplus<<<1,32,0,stream>>>(p_gamma, gsp, HH_);
    k_att<<<dim3((unsigned)(((long long)BB_*HH_*LL_*LL_+255)/256)),256,0,stream>>>(
        qbuf, kvbuf, qpg, kvpg, qn, kn, bz, gsp, mask, att);
    k_softmax<<<dim3(BB_*HH_*LL_),256,0,stream>>>(att);
    k_oout<<<dim3((NRES_*256+255)/256),256,0,stream>>>(att, kvbuf, catbf);
    k_opt<<<dim3((NRES_*576+255)/256),256,0,stream>>>(att, kvpg, optg);
    k_optfin<<<dim3((NRES_*192+255)/256),256,0,stream>>>(optg, Rmat, trans, catbf);
    k_opair<<<dim3((unsigned)(((long long)NRES_*2048+255)/256)),256,0,stream>>>(att, zin, catbf);
    gemm(stream, catbf, wb_wo, p_bo, nullptr, ipaout, NRES_, 256, 3072, 0);
    // dense = LN(res_feats + ipa_out) -> output region 1 (+ bf16 copy)
    k_ln<<<dim3(NRES_),128,0,stream>>>(ipaout, resfeats, p_ipa_g, p_ipa_b,
                                       d_out + OUT_DENSE_OFF, densebf, CS_);

    // ================= Stage D: edge transition =================
    gemm(stream, densebf, wb_et_wi, p_et_bi, nullptr, nebuf, NRES_, NEB_, 256, 0);
    const long long TOTR = (long long)BB_ * LL_ * LL_;       // 131072 rows
    for (long long r0 = 0; r0 < TOTR; r0 += NCH) {
        k_build_eh<<<dim3((unsigned)(((long long)NCH*HID_+255)/256)),256,0,stream>>>(
            zin, nebuf, ehc, r0, NCH);
        gemm(stream, ehc, wb_et_w1, p_et_b1, nullptr, thc, NCH, HID_, HID_,
             GFLAG_RELU | GFLAG_OUTBF);
        gemm(stream, thc, wb_et_w2, p_et_b2, ehc /* residual */, sumc, NCH, HID_, HID_,
             GFLAG_RELU | GFLAG_OUTBF);
        gemm(stream, sumc, wb_et_wf, p_et_bf, nullptr, outc, NCH, CZ_, HID_, 0);
        k_ln<<<dim3(NCH),128,0,stream>>>(outc, nullptr, p_et_g, p_et_b,
                                         d_out + OUT_EDGE_OFF + r0 * CZ_, nullptr, CZ_);
    }
    (void)in_sizes; (void)n_in; (void)out_size; (void)ws_size;
}
